// GraphAttentionNetworkInductive_51170240365251
// MI455X (gfx1250) — matrix operand order repaired and checked by reading, not compiled, not run on hardware
//
#include <hip/hip_runtime.h>
#include <hip/hip_bf16.h>

// Problem constants (from reference)
#define NN   20000
#define EE   320000
#define HIDN 256
#define OUTN 64
#define LLAY 4
#define NHEAD 4
#define UDIM 64

typedef __attribute__((ext_vector_type(8)))  __bf16 v8bf;
typedef __attribute__((ext_vector_type(16))) __bf16 v16bf;
typedef __attribute__((ext_vector_type(8)))  float  v8f;

__device__ __forceinline__ v16bf cat8(v8bf lo, v8bf hi) {
  return __builtin_shufflevector(lo, hi, 0,1,2,3,4,5,6,7,8,9,10,11,12,13,14,15);
}
__device__ __forceinline__ unsigned short f2bf(float f) {
  unsigned u = __float_as_uint(f);
  u += 0x7FFFu + ((u >> 16) & 1u);               // round-to-nearest-even
  return (unsigned short)(u >> 16);
}
// order-preserving float<->uint encoding for atomicMax-based segment max
__device__ __forceinline__ unsigned enc_f32(float f) {
  unsigned u = __float_as_uint(f);
  return (u & 0x80000000u) ? ~u : (u | 0x80000000u);
}
__device__ __forceinline__ float dec_f32(unsigned u) {
  return __uint_as_float((u & 0x80000000u) ? (u ^ 0x80000000u) : ~u);
}
__device__ __forceinline__ float gelu_tanh(float x) {
  float x3 = x * x * x;
  float t = tanhf(0.7978845608028654f * (x + 0.044715f * x3));
  return 0.5f * x * (1.0f + t);
}

// ---------------------------------------------------------------------------
// f32 -> bf16 conversion (activations)
__global__ void __launch_bounds__(256)
cvt_f32_bf16_k(const float* __restrict__ x, unsigned short* __restrict__ y, int n) {
  int i = blockIdx.x * 256 + threadIdx.x;
  if (i < n) y[i] = f2bf(x[i]);
}

// Pack f32 row-major weight [K][M] into per-lane B fragments (bf16):
// Bp[((kb*NB + nb)*32 + lane)*16 + e] = W[kb*32 + (lane>>4)*16 + e][nb*16 + (lane&15)]
__global__ void __launch_bounds__(256)
pack_weight_k(const float* __restrict__ W, unsigned short* __restrict__ Bp, int K, int M) {
  int tid = blockIdx.x * 256 + threadIdx.x;
  int total = K * M;
  if (tid >= total) return;
  int e    = tid & 15;
  int lane = (tid >> 4) & 31;
  int blk  = tid >> 9;
  int NB   = M >> 4;
  int nb   = blk % NB;
  int kb   = blk / NB;
  int k = kb * 32 + (lane >> 4) * 16 + e;
  int n = nb * 16 + (lane & 15);
  Bp[tid] = f2bf(W[(size_t)k * M + n]);
}

// ---------------------------------------------------------------------------
// bf16 WMMA GEMM: C[rows x out_cols] (+bias) from A[rows x K] bf16 row-major
// and packed Bp. Wave tile 32x64 (2x4 fragments of 16x16, K step 32).
// Compile-time epilogue: GELU_ACC ? C += gelu(acc+bias) : C = acc (+bias)
template <bool HAS_BIAS, bool GELU_ACC>
__global__ void __launch_bounds__(256)
gemm_bf16_wmma_k(const unsigned short* __restrict__ A,
                 const unsigned short* __restrict__ Bp,
                 const float* __restrict__ bias,
                 float* __restrict__ C,
                 int K, int NBtot, int nb_off, int out_cols,
                 int total_waves) {
  int wid = blockIdx.x * 8 + (threadIdx.x >> 5);
  if (wid >= total_waves) return;                 // wave-uniform: EXEC stays all-1s
  int lane   = threadIdx.x & 31;
  int lane15 = lane & 15;
  int lhalf  = lane >> 4;                         // 0 or 1
  int ntiles = out_cols >> 6;
  int mt = wid / ntiles, nt = wid % ntiles;
  int row0 = mt * 32, col0 = nt * 64;

  v8f acc[2][4];
  v8f zero = {0.f,0.f,0.f,0.f,0.f,0.f,0.f,0.f};
#pragma unroll
  for (int i = 0; i < 2; ++i)
#pragma unroll
    for (int j = 0; j < 4; ++j) acc[i][j] = zero;

  int KB = K >> 5;
  for (int kb = 0; kb < KB; ++kb) {
    int k0 = kb << 5;
    v16bf afr[2];
#pragma unroll
    for (int mf = 0; mf < 2; ++mf) {
      const unsigned short* ap =
          A + (size_t)(row0 + mf * 16 + lane15) * K + k0 + lhalf * 16;
      v8bf alo = *(const v8bf*)ap;                // K = k0 + lhalf*16 .. +7
      v8bf ahi = *(const v8bf*)(ap + 8);          // K = k0+lhalf*16+8 .. +7
      afr[mf] = cat8(alo, ahi);
    }
    v16bf bfr[4];
#pragma unroll
    for (int nf = 0; nf < 4; ++nf) {
      int nb = nb_off + nt * 4 + nf;
      const unsigned short* bp =
          Bp + (((size_t)kb * NBtot + nb) * 32 + lane) * 16;
      v8bf blo = *(const v8bf*)bp;
      v8bf bhi = *(const v8bf*)(bp + 8);
      bfr[nf] = cat8(blo, bhi);
    }
#pragma unroll
    for (int mf = 0; mf < 2; ++mf)
#pragma unroll
      for (int nf = 0; nf < 4; ++nf)
        acc[mf][nf] = __builtin_amdgcn_wmma_f32_16x16x32_bf16(
            false, afr[mf], false, bfr[nf], (short)0, acc[mf][nf], false, false);
  }

  // Straight-line, branch-free epilogue (all control compile-time).
#pragma unroll
  for (int mf = 0; mf < 2; ++mf)
#pragma unroll
    for (int nf = 0; nf < 4; ++nf) {
      int col = col0 + nf * 16 + lane15;
      float bv = HAS_BIAS ? bias[col] : 0.0f;
      float* cp = C + (size_t)(row0 + mf * 16 + lhalf * 8) * out_cols + col;
#pragma unroll
      for (int v = 0; v < 8; ++v) {               // C layout: lanes16-31 -> M=v+8
        float val = acc[mf][nf][v] + bv;
        if (GELU_ACC) cp[(size_t)v * out_cols] += gelu_tanh(val);
        else          cp[(size_t)v * out_cols]  = val;
      }
    }
}

// ---------------------------------------------------------------------------
// LayerNorm over 256 features, one wave per row
__global__ void __launch_bounds__(256)
layernorm_k(const float* __restrict__ x, const float* __restrict__ g,
            const float* __restrict__ b, float* __restrict__ y, int rows) {
  int wid = blockIdx.x * 8 + (threadIdx.x >> 5);
  if (wid >= rows) return;
  int lane = threadIdx.x & 31;
  const float* xr = x + (size_t)wid * HIDN;
  float v[8], s = 0.f, sq = 0.f;
#pragma unroll
  for (int i = 0; i < 8; ++i) {
    float t = xr[lane + i * 32];
    v[i] = t; s += t; sq += t * t;
  }
#pragma unroll
  for (int o = 16; o > 0; o >>= 1) {
    s  += __shfl_xor(s,  o, 32);
    sq += __shfl_xor(sq, o, 32);
  }
  float mean = s * (1.0f / HIDN);
  float var  = sq * (1.0f / HIDN) - mean * mean;
  float rstd = rsqrtf(var + 1e-3f);
  float* yr = y + (size_t)wid * HIDN;
#pragma unroll
  for (int i = 0; i < 8; ++i) {
    int c = lane + i * 32;
    yr[c] = (v[i] - mean) * rstd * g[c] + b[c];
  }
}

// Per-edge attention logits + segment max. One wave per edge.
__global__ void __launch_bounds__(256)
edge_score_k(const float* __restrict__ zl, const float* __restrict__ zr,
             const int* __restrict__ edges, const float* __restrict__ a,
             float* __restrict__ s_buf, unsigned* __restrict__ smax_enc, int E) {
  int wid = blockIdx.x * 8 + (threadIdx.x >> 5);
  if (wid >= E) return;
  int lane = threadIdx.x & 31;
  int d  = edges[2 * wid];
  int sn = edges[2 * wid + 1];
  const float* zld = zl + (size_t)d  * HIDN;
  const float* zrs = zr + (size_t)sn * HIDN;
#pragma unroll
  for (int h = 0; h < NHEAD; ++h) {
    int u = 2 * lane;
    float f0 = zld[h * UDIM + u]     + zrs[h * UDIM + u];
    float f1 = zld[h * UDIM + u + 1] + zrs[h * UDIM + u + 1];
    f0 = (f0 >= 0.f) ? f0 : 0.2f * f0;
    f1 = (f1 >= 0.f) ? f1 : 0.2f * f1;
    float p = a[h * UDIM + u] * f0 + a[h * UDIM + u + 1] * f1;
#pragma unroll
    for (int o = 16; o > 0; o >>= 1) p += __shfl_xor(p, o, 32);
    if (lane == 0) {
      s_buf[(size_t)wid * NHEAD + h] = p;
      atomicMax(smax_enc + (size_t)d * NHEAD + h, enc_f32(p));
    }
  }
}

// es = exp(s - smax[dst]); denom += es
__global__ void __launch_bounds__(256)
edge_softmax_k(const float* __restrict__ s_buf, const unsigned* __restrict__ smax_enc,
               const int* __restrict__ edges, float* __restrict__ es_buf,
               float* __restrict__ denom, int EH) {
  int i = blockIdx.x * 256 + threadIdx.x;
  if (i >= EH) return;
  int e = i >> 2, h = i & 3;
  int d = edges[2 * e];
  float sm = dec_f32(smax_enc[(size_t)d * NHEAD + h]);
  float es = __expf(s_buf[i] - sm);
  es_buf[i] = es;
  atomicAdd(denom + (size_t)d * NHEAD + h, es);
}

// agg[dst,:] += es * zr[src,:]. One wave per edge, 8 f32 atomics per lane.
__global__ void __launch_bounds__(256)
edge_aggregate_k(const float* __restrict__ zr, const float* __restrict__ es_buf,
                 const int* __restrict__ edges, float* __restrict__ agg, int E) {
  int wid = blockIdx.x * 8 + (threadIdx.x >> 5);
  if (wid >= E) return;
  int lane = threadIdx.x & 31;
  int d  = edges[2 * wid];
  int sn = edges[2 * wid + 1];
  const float* zrs = zr + (size_t)sn * HIDN;
  float* ag = agg + (size_t)d * HIDN;
#pragma unroll
  for (int i = 0; i < 8; ++i) {
    int c = lane + i * 32;
    float w = es_buf[(size_t)wid * NHEAD + (c >> 6)];
    atomicAdd(ag + c, w * zrs[c]);
  }
}

// h = agg/(denom+1e-9) + hn  (residual on the layer-normed input)
__global__ void __launch_bounds__(256)
attn_finalize_k(const float* __restrict__ agg, const float* __restrict__ denom,
                const float* __restrict__ hn, float* __restrict__ hout, int total) {
  int i = blockIdx.x * 256 + threadIdx.x;
  if (i >= total) return;
  int c = i & (HIDN - 1);
  int n = i >> 8;
  float den = denom[(size_t)n * NHEAD + (c >> 6)] + 1e-9f;
  hout[i] = agg[i] / den + hn[i];
}

__global__ void __launch_bounds__(256)
copy_f32_k(const float* __restrict__ x, float* __restrict__ y, int n) {
  int i = blockIdx.x * 256 + threadIdx.x;
  if (i < n) y[i] = x[i];
}

// ---------------------------------------------------------------------------
extern "C" void kernel_launch(void* const* d_in, const int* in_sizes, int n_in,
                              void* d_out, int out_size, void* d_ws, size_t ws_size,
                              hipStream_t stream) {
  (void)in_sizes; (void)n_in; (void)out_size; (void)ws_size;
  const float* x       = (const float*)d_in[0];
  const float* head_W  = (const float*)d_in[1];
  const float* head_b  = (const float*)d_in[2];
  const float* ng_g    = (const float*)d_in[3];
  const float* ng_b    = (const float*)d_in[4];
  const float* Wl      = (const float*)d_in[5];
  const float* Wr      = (const float*)d_in[6];
  const float* att     = (const float*)d_in[7];
  const float* nd_g    = (const float*)d_in[8];
  const float* nd_b    = (const float*)d_in[9];
  const float* dense_W = (const float*)d_in[10];
  const float* dense_b = (const float*)d_in[11];
  const float* tail_W  = (const float*)d_in[12];
  const float* tail_b  = (const float*)d_in[13];
  const int*   edges   = (const int*)d_in[14];
  float* out = (float*)d_out;

  // ---- workspace carve-out
  char* base = (char*)d_ws;
  size_t cur = 0;
  auto carve = [&](size_t bytes) -> void* {
    cur = (cur + 255) & ~(size_t)255;
    void* p = base + cur;
    cur += bytes;
    return p;
  };
  const size_t NH = (size_t)NN * HIDN;
  float*          h_buf   = (float*)carve(NH * 4);
  float*          hn_buf  = (float*)carve(NH * 4);
  float*          zl_buf  = (float*)carve(NH * 4);
  float*          zr_buf  = (float*)carve(NH * 4);
  float*          agg_buf = (float*)carve(NH * 4);
  float*          s_buf   = (float*)carve((size_t)EE * NHEAD * 4);
  float*          es_buf  = (float*)carve((size_t)EE * NHEAD * 4);
  unsigned*       smax    = (unsigned*)carve((size_t)NN * NHEAD * 4);
  float*          denom   = (float*)carve((size_t)NN * NHEAD * 4);
  unsigned short* Abf     = (unsigned short*)carve(NH * 2);
  unsigned short* headWp  = (unsigned short*)carve((size_t)HIDN * HIDN * 2);
  unsigned short* WlP     = (unsigned short*)carve((size_t)LLAY * HIDN * HIDN * 2);
  unsigned short* WrP     = (unsigned short*)carve((size_t)LLAY * HIDN * HIDN * 2);
  unsigned short* denseWp = (unsigned short*)carve((size_t)LLAY * HIDN * 4 * HIDN * 2);
  unsigned short* tailWp  = (unsigned short*)carve((size_t)HIDN * OUTN * 2);

  // mode: 0 = plain no bias, 1 = plain + bias, 2 = gelu-accumulate + bias
  auto gemm = [&](const unsigned short* A, const unsigned short* Bp,
                  const float* bias, float* C, int K, int NBtot, int nb_off,
                  int out_cols, int mode) {
    int waves = (NN / 32) * (out_cols / 64);
    int blocks = (waves + 7) / 8;
    if (mode == 0)
      gemm_bf16_wmma_k<false, false><<<blocks, 256, 0, stream>>>(
          A, Bp, bias, C, K, NBtot, nb_off, out_cols, waves);
    else if (mode == 1)
      gemm_bf16_wmma_k<true, false><<<blocks, 256, 0, stream>>>(
          A, Bp, bias, C, K, NBtot, nb_off, out_cols, waves);
    else
      gemm_bf16_wmma_k<true, true><<<blocks, 256, 0, stream>>>(
          A, Bp, bias, C, K, NBtot, nb_off, out_cols, waves);
  };

  // ---- pack all weights to bf16 fragment layout
  pack_weight_k<<<(HIDN * HIDN + 255) / 256, 256, 0, stream>>>(head_W, headWp, HIDN, HIDN);
  for (int l = 0; l < LLAY; ++l) {
    pack_weight_k<<<(HIDN * HIDN + 255) / 256, 256, 0, stream>>>(
        Wl + (size_t)l * HIDN * HIDN, WlP + (size_t)l * HIDN * HIDN, HIDN, HIDN);
    pack_weight_k<<<(HIDN * HIDN + 255) / 256, 256, 0, stream>>>(
        Wr + (size_t)l * HIDN * HIDN, WrP + (size_t)l * HIDN * HIDN, HIDN, HIDN);
    pack_weight_k<<<(HIDN * 4 * HIDN + 255) / 256, 256, 0, stream>>>(
        dense_W + (size_t)l * HIDN * 4 * HIDN,
        denseWp + (size_t)l * HIDN * 4 * HIDN, HIDN, 4 * HIDN);
  }
  pack_weight_k<<<(HIDN * OUTN + 255) / 256, 256, 0, stream>>>(tail_W, tailWp, HIDN, OUTN);

  const int elemN  = (int)NH;
  const int cvtBlk = (elemN + 255) / 256;

  // ---- head: h = x @ head_W + head_b
  cvt_f32_bf16_k<<<cvtBlk, 256, 0, stream>>>(x, Abf, elemN);
  gemm(Abf, headWp, head_b, h_buf, HIDN, HIDN / 16, 0, HIDN, 1);

  for (int l = 0; l < LLAY; ++l) {
    // hn = LN(h)
    layernorm_k<<<(NN + 7) / 8, 256, 0, stream>>>(h_buf, ng_g + l * HIDN,
                                                  ng_b + l * HIDN, hn_buf, NN);
    cvt_f32_bf16_k<<<cvtBlk, 256, 0, stream>>>(hn_buf, Abf, elemN);
    // zl, zr
    gemm(Abf, WlP + (size_t)l * HIDN * HIDN, nullptr, zl_buf, HIDN, HIDN / 16, 0, HIDN, 0);
    gemm(Abf, WrP + (size_t)l * HIDN * HIDN, nullptr, zr_buf, HIDN, HIDN / 16, 0, HIDN, 0);
    // attention softmax + aggregation
    hipMemsetAsync(agg_buf, 0, NH * 4, stream);
    hipMemsetAsync(smax, 0, (size_t)NN * NHEAD * 4, stream);   // enc identity = -inf
    hipMemsetAsync(denom, 0, (size_t)NN * NHEAD * 4, stream);
    edge_score_k<<<(EE + 7) / 8, 256, 0, stream>>>(zl_buf, zr_buf, edges,
                                                   att + l * NHEAD * UDIM, s_buf, smax, EE);
    edge_softmax_k<<<(EE * NHEAD + 255) / 256, 256, 0, stream>>>(s_buf, smax, edges,
                                                                 es_buf, denom, EE * NHEAD);
    edge_aggregate_k<<<(EE + 7) / 8, 256, 0, stream>>>(zr_buf, es_buf, edges, agg_buf, EE);
    attn_finalize_k<<<cvtBlk, 256, 0, stream>>>(agg_buf, denom, hn_buf, h_buf, elemN);
    // xr = LN(h); h = xr + sum_j gelu(xr @ denseW[:, j*HID:(j+1)*HID] + b_j)
    layernorm_k<<<(NN + 7) / 8, 256, 0, stream>>>(h_buf, nd_g + l * HIDN,
                                                  nd_b + l * HIDN, hn_buf, NN);
    cvt_f32_bf16_k<<<cvtBlk, 256, 0, stream>>>(hn_buf, Abf, elemN);
    copy_f32_k<<<cvtBlk, 256, 0, stream>>>(hn_buf, h_buf, elemN);
    for (int j = 0; j < 4; ++j) {
      gemm(Abf, denseWp + (size_t)l * HIDN * 4 * HIDN,
           dense_b + (size_t)l * 4 * HIDN + j * HIDN, h_buf,
           HIDN, (4 * HIDN) / 16, j * (HIDN / 16), HIDN, 2);
    }
  }

  // ---- tail: out = h @ tail_W + tail_b
  cvt_f32_bf16_k<<<cvtBlk, 256, 0, stream>>>(h_buf, Abf, elemN);
  gemm(Abf, tailWp, tail_b, out, HIDN, OUTN / 16, 0, OUTN, 1);
}
